// sig_t_mean_32272384262399
// MI455X (gfx1250) — matrix operand order repaired and checked by reading, not compiled, not run on hardware
//
#include <hip/hip_runtime.h>
#include <hip/hip_bf16.h>

typedef __attribute__((ext_vector_type(16))) _Float16 v16h;
typedef __attribute__((ext_vector_type(8)))  _Float16 v8h;
typedef __attribute__((ext_vector_type(4)))  _Float16 v4h;
typedef __attribute__((ext_vector_type(2)))  _Float16 v2h;
typedef __attribute__((ext_vector_type(8)))  float    v8f;
typedef __attribute__((ext_vector_type(4)))  float    v4f;

#define EPSV 1e-5f

// Problem constants (from reference: B=4, C=64, N=16384)
#define CDIM   64
#define NDIM   16384
#define TM     256           // rows per block (each of 8 waves: 32 rows)
#define PITCH  72            // halves per LDS row (144B stride: 16B aligned)

__global__ __launch_bounds__(256) void sig_t_mean_kernel(
    const float* __restrict__ x,    // (B, C, N)
    const float* __restrict__ cm,   // (C, C)
    const float* __restrict__ W,    // (C, C, 2C)
    float* __restrict__ out)        // (M, C*C) = (65536, 4096)
{
    __shared__ __attribute__((aligned(16))) _Float16 lds_a[TM * PITCH];   // 36.0 KB
    __shared__ __attribute__((aligned(16))) _Float16 lds_w[CDIM * PITCH]; //  9.0 KB
    __shared__ float lds_bias[CDIM];

    const int tid = threadIdx.x;
    const int k   = blockIdx.y;               // 0..63
    const int m0  = blockIdx.x * TM;          // row block
    const int b   = m0 / NDIM;
    const int n0  = m0 % NDIM;

    const float* xb = x + (size_t)b * CDIM * NDIM + n0;
    const float* wk = W + (size_t)k * CDIM * 2 * CDIM;   // W[k, i, j], i-stride = 2C

    // ---- Phase 0a: stage A tile (TM x C) into LDS as f16 (transposed from x) ----
    // Unit = 2 columns (j, j+1) x 4 rows (r..r+3): two v4f loads along contiguous n,
    // packed-convert into four b32 LDS stores.
    #pragma unroll
    for (int u = tid; u < (CDIM / 2) * (TM / 4); u += 256) {   // 2048 units, 8/thread
        int jp = u >> 6;                 // 0..31
        int rq = u & 63;                 // 0..63
        int j  = jp * 2;
        int r  = rq * 4;
        v4f a0 = *(const v4f*)(xb + (size_t)j * NDIM + r);
        v4f a1 = *(const v4f*)(xb + (size_t)(j + 1) * NDIM + r);
        #pragma unroll
        for (int q = 0; q < 4; ++q) {
            v2h p;
            p[0] = (_Float16)a0[q];
            p[1] = (_Float16)a1[q];
            *(v2h*)&lds_a[(r + q) * PITCH + j] = p;
        }
    }

    // ---- Phase 0b: stage Wx slice (C x C) into LDS as f16 (shared by all waves) ----
    #pragma unroll
    for (int u = tid; u < CDIM * (CDIM / 4); u += 256) {       // 1024 units, 4/thread
        int i = u >> 4;
        int j = (u & 15) * 4;
        v4f wv = *(const v4f*)(wk + i * 2 * CDIM + j);
        v4h p;
        #pragma unroll
        for (int q = 0; q < 4; ++q) p[q] = (_Float16)wv[q];
        *(v4h*)&lds_w[i * PITCH + j] = p;
    }

    // ---- Phase 0c: bias[i] = sum_j cm[k,j] * W[k,i,64+j] ----
    if (tid < CDIM) {
        const float* wr = wk + (size_t)tid * 2 * CDIM + CDIM;
        const float* cr = cm + (size_t)k * CDIM;
        float s = 0.f;
        #pragma unroll
        for (int j4 = 0; j4 < CDIM / 4; ++j4) {
            v4f wv = *(const v4f*)(wr + j4 * 4);
            s += cr[j4 * 4 + 0] * wv[0] + cr[j4 * 4 + 1] * wv[1]
               + cr[j4 * 4 + 2] * wv[2] + cr[j4 * 4 + 3] * wv[3];
        }
        lds_bias[tid] = s;
    }
    __syncthreads();

    const int wave = tid >> 5;     // 0..7 -> 32-row slab
    const int lane = tid & 31;
    const int lh   = lane >> 4;    // lane half
    const int ln   = lane & 15;

    // ---- Accumulators: 2 row-frags x 4 col-tiles of 16x16; preload bias ----
    v8f acc[2][4];
    #pragma unroll
    for (int t = 0; t < 4; ++t) {
        float bv = lds_bias[t * 16 + ln];    // column i = t*16 + ln (same for all rows)
        #pragma unroll
        for (int rf = 0; rf < 2; ++rf)
            #pragma unroll
            for (int e = 0; e < 8; ++e) acc[rf][t][e] = bv;
    }

    // ---- GEMM: K = 64 = 2 chunks of 32 ----
    #pragma unroll
    for (int kk = 0; kk < 2; ++kk) {
        // A fragments (16x32 f16): lane half lh owns K in {lh*8..+7} U {16+lh*8..+7}
        v16h afrag[2];
        #pragma unroll
        for (int rf = 0; rf < 2; ++rf) {
            const int rowL = wave * 32 + rf * 16 + ln;
            const _Float16* abase = &lds_a[rowL * PITCH + kk * 32 + lh * 8];
            v8h alo = *(const v8h*)(abase);        // K = lh*8 + 0..7
            v8h ahi = *(const v8h*)(abase + 16);   // K = 16 + lh*8 + 0..7
            #pragma unroll
            for (int e = 0; e < 8; ++e) { afrag[rf][e] = alo[e]; afrag[rf][e + 8] = ahi[e]; }
        }

        #pragma unroll
        for (int t = 0; t < 4; ++t) {
            // B fragment (32x16 f16): lane half lh owns K in {lh*8..+7} U {16+lh*8..+7}
            const int i = t * 16 + ln;
            const _Float16* wbase = &lds_w[i * PITCH + kk * 32 + lh * 8];
            v8h b0 = *(const v8h*)(wbase);
            v8h b1 = *(const v8h*)(wbase + 16);
            v16h bfrag;
            #pragma unroll
            for (int e = 0; e < 8; ++e) { bfrag[e] = b0[e]; bfrag[e + 8] = b1[e]; }

            #pragma unroll
            for (int rf = 0; rf < 2; ++rf) {
                acc[rf][t] = __builtin_amdgcn_wmma_f32_16x16x32_f16(
                    /*neg_a=*/false, afrag[rf], /*neg_b=*/false, bfrag,
                    /*c_mod=*/(short)0, acc[rf][t], /*reuse_a=*/false, /*reuse_b=*/false);
            }
        }
    }

    // ---- Epilogue per row-fragment: clip, 64-wide row sum, normalize, store ----
    #pragma unroll
    for (int rf = 0; rf < 2; ++rf) {
        // clip to [EPS, 1-EPS] in one med3 per element (values are finite)
        #pragma unroll
        for (int t = 0; t < 4; ++t)
            #pragma unroll
            for (int e = 0; e < 8; ++e)
                acc[rf][t][e] = __builtin_amdgcn_fmed3f(acc[rf][t][e], EPSV, 1.0f - EPSV);

        // Row of element e lives in lane-half lh (M = e + lh*8); reduce over the
        // 16 lanes of that half plus the 4 tiles held in-thread.
        float inv[8];
        #pragma unroll
        for (int e = 0; e < 8; ++e) {
            float s = acc[rf][0][e] + acc[rf][1][e] + acc[rf][2][e] + acc[rf][3][e];
            s += __shfl_xor(s, 1, 16);
            s += __shfl_xor(s, 2, 16);
            s += __shfl_xor(s, 4, 16);
            s += __shfl_xor(s, 8, 16);
            inv[e] = 1.0f / s;
        }

        // Store: out[m, k*64 + i] with non-temporal hint (write-once stream)
        const int mbase = m0 + wave * 32 + rf * 16 + lh * 8;
        #pragma unroll
        for (int t = 0; t < 4; ++t) {
            const int col = k * 64 + t * 16 + ln;
            #pragma unroll
            for (int e = 0; e < 8; ++e) {
                size_t off = (size_t)(mbase + e) * (CDIM * CDIM) + col;
                __builtin_nontemporal_store(acc[rf][t][e] * inv[e], &out[off]);
            }
        }
    }
}

extern "C" void kernel_launch(void* const* d_in, const int* in_sizes, int n_in,
                              void* d_out, int out_size, void* d_ws, size_t ws_size,
                              hipStream_t stream) {
    const float* x  = (const float*)d_in[0];   // (4, 64, 16384)
    const float* cm = (const float*)d_in[1];   // (64, 64)
    const float* W  = (const float*)d_in[2];   // (64, 64, 128)
    float* out = (float*)d_out;                // (65536, 4096)

    const int M = (4 * NDIM);                  // 65536 rows
    dim3 grid(M / TM, CDIM);                   // (256, 64)
    dim3 block(256);
    sig_t_mean_kernel<<<grid, block, 0, stream>>>(x, cm, W, out);
    (void)in_sizes; (void)n_in; (void)out_size; (void)d_ws; (void)ws_size;
}
